// CopyDecoder_49727131353647
// MI455X (gfx1250) — matrix operand order repaired and checked by reading, not compiled, not run on hardware
//
#include <hip/hip_runtime.h>
#include <hip/hip_bf16.h>
#include <cstddef>

// ---------------- problem constants ----------------
#define V     50000
#define E     256
#define H     512
#define B     32
#define S     512
#define TWOH  1024          // 2*H
#define XDIM  1280          // E + 2H
#define G3    1536          // 3*H

// A-panel LDS: 16 rows x 1024 bf16, padded to 1040 shorts/row (anti-conflict)
#define AROW  1040

// ---------------- workspace layout (bytes) ----------------
#define OFF_X     0                               // x:      B*XDIM f32   = 163840
#define OFF_HEFF  163840                          // h_eff:  B*H f32      = 65536
#define OFF_ST    229376                          // state:  B*H f32      = 65536
#define OFF_SC    294912                          // score_c:B*S f32      = 65536
#define OFF_BMAX  360448                          // bmax:   B f32 (pad 256)
#define OFF_RSUM  360704                          // rsum:   B f32 (pad 256)
#define OFF_WB    360960                          // Wc bf16 swizzled: H*TWOH*2 = 1048576

// out layout: [B*V | B*H | B*TWOH]
#define OUT_STATE (B * V)
#define OUT_WNEW  (B * V + B * H)

typedef __attribute__((ext_vector_type(16))) __bf16        v16bf;
typedef __attribute__((ext_vector_type(8)))  float         v8f;
typedef __attribute__((ext_vector_type(8)))  unsigned short u16x8;
typedef __attribute__((ext_vector_type(2)))  unsigned int   u32x2;

__device__ __forceinline__ unsigned int pack2_bf16(float a, float b) {
  unsigned int ua = __float_as_uint(a);
  unsigned int ub = __float_as_uint(b);
  ua = ua + 0x7FFFu + ((ua >> 16) & 1u);   // round-to-nearest-even
  ub = ub + 0x7FFFu + ((ub >> 16) & 1u);
  return (ua >> 16) | (ub & 0xFFFF0000u);
}

__device__ __forceinline__ float sigmoidf(float x) {
  return 1.0f / (1.0f + __expf(-x));
}

// ---------------------------------------------------------------------------
// Kernel 0: swizzle Wc_w (H x 2H, f32) into WMMA-B fragment order, bf16.
// Fragment (ht,kt): lane l holds n=l%16, k = kt*32 + 16*(l/16) + e, e=0..15.
// Stored fragment-major: 16 shorts per lane, 32 lanes per fragment => each
// lane's fragment is one contiguous, 32B-aligned chunk.
// ---------------------------------------------------------------------------
__global__ __launch_bounds__(256) void k_swizzle_wc(const float* __restrict__ Wc_w,
                                                    unsigned short* __restrict__ wB) {
  int t = blockIdx.x * blockDim.x + threadIdx.x;      // (tile, lane)
  if (t >= (H / 16) * (TWOH / 32) * 32) return;
  int lane = t & 31;
  int tile = t >> 5;
  int ht = tile >> 5;            // h-tile 0..31
  int kt = tile & 31;            // k-tile 0..31
  int n  = lane & 15;
  int hs = lane >> 4;
  const float* src = Wc_w + (size_t)(ht * 16 + n) * TWOH + kt * 32 + hs * 16;
  unsigned int* dst = (unsigned int*)(wB + (size_t)t * 16);
#pragma unroll
  for (int e = 0; e < 16; e += 2)
    dst[e >> 1] = pack2_bf16(src[e], src[e + 1]);
}

// ---------------------------------------------------------------------------
// Kernel 1: h_eff = (order==0) ? encoded[:,0]@Ws_w.T + Ws_b : prev_state
// ---------------------------------------------------------------------------
__global__ __launch_bounds__(256) void k_heff(const float* __restrict__ encoded,
                                              const float* __restrict__ prev_state,
                                              const float* __restrict__ Ws_w,
                                              const float* __restrict__ Ws_b,
                                              const int* __restrict__ order_p,
                                              float* __restrict__ heff) {
  int t = blockIdx.x * blockDim.x + threadIdx.x;
  if (t >= B * H) return;
  int b = t & 31;
  int h = t >> 5;
  if (order_p[0] == 0) {
    const float* er = encoded + (size_t)b * S * TWOH;     // encoded[b,0,:]
    const float* wr = Ws_w + (size_t)h * TWOH;
    float acc = Ws_b[h];
    const float4* e4 = (const float4*)er;
    const float4* w4 = (const float4*)wr;
    for (int k = 0; k < TWOH / 4; ++k) {
      float4 a = e4[k], w = w4[k];
      acc += a.x * w.x + a.y * w.y + a.z * w.z + a.w * w.w;
    }
    heff[b * H + h] = acc;
  } else {
    heff[b * H + h] = prev_state[b * H + h];
  }
}

// ---------------------------------------------------------------------------
// Kernel 2: x = concat(embed_w[input_idx], weighted_eff)
// ---------------------------------------------------------------------------
__global__ __launch_bounds__(256) void k_buildx(const int* __restrict__ input_idx,
                                                const float* __restrict__ embed_w,
                                                const float* __restrict__ weighted,
                                                const int* __restrict__ order_p,
                                                float* __restrict__ x) {
  int t = blockIdx.x * blockDim.x + threadIdx.x;
  if (t >= B * XDIM) return;
  int b = t / XDIM;
  int c = t % XDIM;
  float v;
  if (c < E)
    v = embed_w[(size_t)input_idx[b] * E + c];
  else
    v = (order_p[0] == 0) ? 0.0f : weighted[b * TWOH + (c - E)];
  x[t] = v;
}

// ---------------------------------------------------------------------------
// Kernel 3: GRU cell -> state (ws) and out[OUT_STATE..]
// ---------------------------------------------------------------------------
__global__ __launch_bounds__(256) void k_gru(const float* __restrict__ x,
                                             const float* __restrict__ heff,
                                             const float* __restrict__ wih,
                                             const float* __restrict__ whh,
                                             const float* __restrict__ bih,
                                             const float* __restrict__ bhh,
                                             float* __restrict__ state,
                                             float* __restrict__ state_out) {
  int t = blockIdx.x * blockDim.x + threadIdx.x;
  if (t >= B * H) return;
  int b = t & 31;              // lanes span batch -> weight rows are wave-uniform
  int h = t >> 5;
  const float4* x4 = (const float4*)(x + (size_t)b * XDIM);
  const float4* h4 = (const float4*)(heff + (size_t)b * H);
  float gi[3], gh[3];
#pragma unroll
  for (int g = 0; g < 3; ++g) {
    int row = g * H + h;
    const float4* wi4 = (const float4*)(wih + (size_t)row * XDIM);
    float acc = bih[row];
    for (int k = 0; k < XDIM / 4; ++k) {
      float4 a = x4[k], w = wi4[k];
      acc += a.x * w.x + a.y * w.y + a.z * w.z + a.w * w.w;
    }
    gi[g] = acc;
    const float4* wh4 = (const float4*)(whh + (size_t)row * H);
    float acch = bhh[row];
    for (int k = 0; k < H / 4; ++k) {
      float4 a = h4[k], w = wh4[k];
      acch += a.x * w.x + a.y * w.y + a.z * w.z + a.w * w.w;
    }
    gh[g] = acch;
  }
  float r = sigmoidf(gi[0] + gh[0]);
  float z = sigmoidf(gi[1] + gh[1]);
  float n = tanhf(gi[2] + r * gh[2]);
  float hp = heff[b * H + h];
  float st = (1.0f - z) * n + z * hp;
  state[b * H + h] = st;
  state_out[b * H + h] = st;
}

// ---------------------------------------------------------------------------
// Kernel 4: score_g = state @ Wo_w.T + Wo_b -> staged into out[0..B*V)
// HBM-bound on Wo_w (102 MB); weight row is wave-uniform => scalar loads.
// ---------------------------------------------------------------------------
__global__ __launch_bounds__(256) void k_score_g(const float* __restrict__ state,
                                                 const float* __restrict__ Wo_w,
                                                 const float* __restrict__ Wo_b,
                                                 float* __restrict__ out) {
  int t = blockIdx.x * blockDim.x + threadIdx.x;
  if (t >= B * V) return;
  int b = t & 31;
  int v = t >> 5;
  const float4* w4 = (const float4*)(Wo_w + (size_t)v * H);
  const float4* s4 = (const float4*)(state + (size_t)b * H);
  float acc = Wo_b[v];
  for (int k = 0; k < H / 4; ++k) {
    float4 a = s4[k], w = w4[k];
    acc += a.x * w.x + a.y * w.y + a.z * w.z + a.w * w.w;
  }
  out[(size_t)b * V + v] = acc;
}

// ---------------------------------------------------------------------------
// Kernel 5 (WMMA): score_c[b,s] = sum_h sigmoid(encoded[b,s,:]@Wc_w[h,:] + Wc_b[h]) * state[b,h]
// One wave per (b, 16-row s-tile). A panel (16x1024) converted f32->bf16 in LDS.
// B fragments read pre-swizzled from ws. 32 h-tiles x 32 k-steps of
// v_wmma_f32_16x16x32_bf16, fused sigmoid + state-weighted row reduction.
// ---------------------------------------------------------------------------
__global__ __launch_bounds__(32) void k_score_c_wmma(const float* __restrict__ encoded,
                                                     const unsigned short* __restrict__ wB,
                                                     const float* __restrict__ state,
                                                     const float* __restrict__ Wc_b,
                                                     float* __restrict__ score_c) {
  __shared__ unsigned short At[16 * AROW];          // 33,280 B
  const int lane = threadIdx.x;
  const int stile = blockIdx.x;                     // 0..31
  const int b = blockIdx.y;                         // 0..31

  // Load 16x1024 f32 panel, convert to bf16 rows in LDS.
  const float* panel = encoded + ((size_t)(b * S + stile * 16)) * TWOH;
  for (int i = lane; i < 16 * (TWOH / 4); i += 32) {
    int row = i >> 8;                 // / (TWOH/4)
    int c4 = (i & 255) << 2;
    float4 f = *(const float4*)(panel + (size_t)row * TWOH + c4);
    u32x2 p;
    p.x = pack2_bf16(f.x, f.y);
    p.y = pack2_bf16(f.z, f.w);
    *(u32x2*)(&At[row * AROW + c4]) = p;
  }
  __syncthreads();

  const int n = lane & 15;            // A row m / D col n
  const int hs = lane >> 4;

  float sc[8];
#pragma unroll
  for (int r = 0; r < 8; ++r) sc[r] = 0.0f;

  for (int ht = 0; ht < 32; ++ht) {
    float bias = Wc_b[ht * 16 + n];
    v8f acc;
#pragma unroll
    for (int r = 0; r < 8; ++r) acc[r] = bias;

    const unsigned short* bp = wB + ((size_t)(ht * 32) * 32 + lane) * 16;
    for (int kt = 0; kt < 32; ++kt) {
      // A fragment: k = kt*32 + 16*hs + e, e=0..15
      const unsigned short* ap = &At[n * AROW + kt * 32 + hs * 16];
      struct __align__(32) AF { u16x8 lo, hi; } af;
      af.lo = *(const u16x8*)(ap);
      af.hi = *(const u16x8*)(ap + 8);
      v16bf a = __builtin_bit_cast(v16bf, af);
      // B fragment: contiguous 32B per lane from swizzled buffer
      struct __align__(32) BF { u16x8 lo, hi; } bf;
      bf.lo = *(const u16x8*)(bp);
      bf.hi = *(const u16x8*)(bp + 8);
      v16bf bb = __builtin_bit_cast(v16bf, bf);
      acc = __builtin_amdgcn_wmma_f32_16x16x32_bf16(
          /*neg_a=*/false, a, /*neg_b=*/false, bb,
          /*c_mod=*/(short)0, acc, /*reuse_a=*/false, /*reuse_b=*/false);
      bp += 32 * 16;                  // next k-tile fragment
    }
    float sv = state[b * H + ht * 16 + n];
#pragma unroll
    for (int r = 0; r < 8; ++r)
      sc[r] += sigmoidf(acc[r]) * sv;
  }

  // Reduce over the 16 n-lanes of each half-wave.
#pragma unroll
  for (int r = 0; r < 8; ++r) {
    float vr = sc[r];
    vr += __shfl_xor(vr, 1, 16);
    vr += __shfl_xor(vr, 2, 16);
    vr += __shfl_xor(vr, 4, 16);
    vr += __shfl_xor(vr, 8, 16);
    sc[r] = vr;
  }
  if (n == 0) {
#pragma unroll
    for (int r = 0; r < 8; ++r)
      score_c[b * S + stile * 16 + r + 8 * hs] = sc[r];   // row m = r + 8*hs
  }
}

// ---------------------------------------------------------------------------
// Kernel 6: per-batch softmax stats over [score_g | score_c] (max, 1/sum)
// ---------------------------------------------------------------------------
__global__ __launch_bounds__(256) void k_stats(const float* __restrict__ out,
                                               const float* __restrict__ score_c,
                                               float* __restrict__ bmax,
                                               float* __restrict__ rsum) {
  __shared__ float red[256];
  int b = blockIdx.x, tid = threadIdx.x;
  float m = -3.402823466e38f;
  for (int i = tid; i < V; i += 256) m = fmaxf(m, out[(size_t)b * V + i]);
  for (int i = tid; i < S; i += 256) m = fmaxf(m, score_c[b * S + i]);
  red[tid] = m;
  __syncthreads();
  for (int s = 128; s > 0; s >>= 1) {
    if (tid < s) red[tid] = fmaxf(red[tid], red[tid + s]);
    __syncthreads();
  }
  float bm = red[0];
  __syncthreads();
  float sum = 0.0f;
  for (int i = tid; i < V; i += 256) sum += __expf(out[(size_t)b * V + i] - bm);
  for (int i = tid; i < S; i += 256) sum += __expf(score_c[b * S + i] - bm);
  red[tid] = sum;
  __syncthreads();
  for (int s = 128; s > 0; s >>= 1) {
    if (tid < s) red[tid] += red[tid + s];
    __syncthreads();
  }
  if (tid == 0) {
    bmax[b] = bm;
    rsum[b] = 1.0f / red[0];
  }
}

// ---------------------------------------------------------------------------
// Kernel 7: in-place prob_g = exp(score_g - max) * rsum
// ---------------------------------------------------------------------------
__global__ __launch_bounds__(256) void k_probg(float* __restrict__ out,
                                               const float* __restrict__ bmax,
                                               const float* __restrict__ rsum) {
  int t = blockIdx.x * blockDim.x + threadIdx.x;
  if (t >= B * V) return;
  int b = t / V;
  out[t] = __expf(out[t] - bmax[b]) * rsum[b];
}

// ---------------------------------------------------------------------------
// Kernel 8: scatter-add prob_c into out rows by encoded_idx
// ---------------------------------------------------------------------------
__global__ __launch_bounds__(256) void k_scatter(const float* __restrict__ score_c,
                                                 const int* __restrict__ encoded_idx,
                                                 const float* __restrict__ bmax,
                                                 const float* __restrict__ rsum,
                                                 float* __restrict__ out) {
  int t = blockIdx.x * blockDim.x + threadIdx.x;
  if (t >= B * S) return;
  int b = t >> 9;
  float p = __expf(score_c[t] - bmax[b]) * rsum[b];
  atomicAdd(&out[(size_t)b * V + encoded_idx[t]], p);
}

// ---------------------------------------------------------------------------
// Kernel 9: weighted_new = (normalized masked score_c) @ encoded  (per batch)
// ---------------------------------------------------------------------------
__global__ __launch_bounds__(256) void k_weighted(const float* __restrict__ score_c,
                                                  const int* __restrict__ encoded_idx,
                                                  const int* __restrict__ input_idx,
                                                  const float* __restrict__ encoded,
                                                  float* __restrict__ outw) {
  __shared__ float sw[S];
  __shared__ float red[256];
  int b = blockIdx.x, tid = threadIdx.x;
  int ii = input_idx[b];
  for (int s = tid; s < S; s += 256) {
    int idx = encoded_idx[b * S + s];
    float v = score_c[b * S + s];
    sw[s] = (idx != 0 && idx == ii) ? v : 0.0f;
  }
  __syncthreads();
  float part = 0.0f;
  for (int s = tid; s < S; s += 256) part += sw[s];
  red[tid] = part;
  __syncthreads();
  for (int s = 128; s > 0; s >>= 1) {
    if (tid < s) red[tid] += red[tid + s];
    __syncthreads();
  }
  float denom = red[0];
  float scale = denom > 0.0f ? 1.0f / denom : 1.0f;
  for (int d = tid; d < TWOH; d += 256) {
    float acc = 0.0f;
    for (int s = 0; s < S; ++s) {
      float w = sw[s];
      if (w != 0.0f) acc += w * encoded[((size_t)(b * S + s)) * TWOH + d];
    }
    outw[b * TWOH + d] = acc * scale;
  }
}

// ---------------------------------------------------------------------------
extern "C" void kernel_launch(void* const* d_in, const int* in_sizes, int n_in,
                              void* d_out, int out_size, void* d_ws, size_t ws_size,
                              hipStream_t stream) {
  (void)in_sizes; (void)n_in; (void)out_size; (void)ws_size;
  const int*   input_idx   = (const int*)d_in[0];
  const float* encoded     = (const float*)d_in[1];
  const int*   encoded_idx = (const int*)d_in[2];
  const float* prev_state  = (const float*)d_in[3];
  const float* weighted    = (const float*)d_in[4];
  const int*   order       = (const int*)d_in[5];
  const float* embed_w     = (const float*)d_in[6];
  const float* gru_wih     = (const float*)d_in[7];
  const float* gru_whh     = (const float*)d_in[8];
  const float* gru_bih     = (const float*)d_in[9];
  const float* gru_bhh     = (const float*)d_in[10];
  const float* Wo_w        = (const float*)d_in[11];
  const float* Wo_b        = (const float*)d_in[12];
  const float* Wc_w        = (const float*)d_in[13];
  const float* Wc_b        = (const float*)d_in[14];
  const float* Ws_w        = (const float*)d_in[15];
  const float* Ws_b        = (const float*)d_in[16];

  float* out = (float*)d_out;
  char*  ws  = (char*)d_ws;
  float* x       = (float*)(ws + OFF_X);
  float* heff    = (float*)(ws + OFF_HEFF);
  float* state   = (float*)(ws + OFF_ST);
  float* score_c = (float*)(ws + OFF_SC);
  float* bmax    = (float*)(ws + OFF_BMAX);
  float* rsum    = (float*)(ws + OFF_RSUM);
  unsigned short* wB = (unsigned short*)(ws + OFF_WB);

  k_swizzle_wc<<<128, 256, 0, stream>>>(Wc_w, wB);
  k_heff<<<(B * H + 255) / 256, 256, 0, stream>>>(encoded, prev_state, Ws_w, Ws_b, order, heff);
  k_buildx<<<(B * XDIM + 255) / 256, 256, 0, stream>>>(input_idx, embed_w, weighted, order, x);
  k_gru<<<(B * H + 255) / 256, 256, 0, stream>>>(x, heff, gru_wih, gru_whh, gru_bih, gru_bhh,
                                                 state, out + OUT_STATE);
  k_score_g<<<(B * V + 255) / 256, 256, 0, stream>>>(state, Wo_w, Wo_b, out);
  k_score_c_wmma<<<dim3(S / 16, B), 32, 0, stream>>>(encoded, wB, state, Wc_b, score_c);
  k_stats<<<B, 256, 0, stream>>>(out, score_c, bmax, rsum);
  k_probg<<<(B * V + 255) / 256, 256, 0, stream>>>(out, bmax, rsum);
  k_scatter<<<(B * S + 255) / 256, 256, 0, stream>>>(score_c, encoded_idx, bmax, rsum, out);
  k_weighted<<<B, 256, 0, stream>>>(score_c, encoded_idx, input_idx, encoded, out + OUT_WNEW);
}
